// LowRankTokenMixing_43748536877648
// MI455X (gfx1250) — matrix operand order repaired and checked by reading, not compiled, not run on hardware
//
#include <hip/hip_runtime.h>

// Causal attention: O = softmax(Q K^T / sqrt(64), causal) * V
// Q=C (d_in[2]), K=B (d_in[1]), V=x (d_in[0]); shapes [2,8,4096,64] fp32.
// Flash-attention, bf16 WMMA, transposed second GEMM (O^T = V^T P^T) so the
// softmax state is per-lane scalar. This round: double-buffered K/V staging,
// diagonal-only masking fast path, hw bf16 pack, base-2 softmax (v_exp_f32).

#define L_SEQ    4096
#define D_HEAD   64
#define QTILE    128     // query rows per block (8 waves x 16 rows)
#define KTILE    32      // keys per iteration
#define NTHREADS 256

// 1/sqrt(64) * log2(e): scores scaled once into the base-2 domain
#define SCALE_LOG2E 0.1803368801111244f

typedef __attribute__((ext_vector_type(16))) __bf16 v16bf;
typedef __attribute__((ext_vector_type(2)))  __bf16 v2bf;
typedef __attribute__((ext_vector_type(8)))  float  v8f;

union Frag16 {
  v16bf bf;
  unsigned int u[8];
};

#if __has_builtin(__builtin_amdgcn_exp2f)
#define EXP2(x) __builtin_amdgcn_exp2f(x)
#else
#define EXP2(x) exp2f(x)
#endif

__device__ __forceinline__ unsigned int bfpack(float a, float b) {
#if __has_builtin(__builtin_amdgcn_cvt_pk_bf16_f32)
  v2bf t = __builtin_amdgcn_cvt_pk_bf16_f32(a, b);
  return *(unsigned int*)&t;
#else
  union { v2bf h; unsigned int u; } t;
  t.h[0] = (__bf16)a; t.h[1] = (__bf16)b;
  return t.u;
#endif
}

// ds_swizzle_b32, group-of-32, xor_mask=0x10 (SWAPX16): lane <-> lane^16
__device__ __forceinline__ float swap16_f(float v) {
  return __int_as_float(__builtin_amdgcn_ds_swizzle(__float_as_int(v), 0x401f));
}

__device__ __forceinline__ void stage_kv(const float* __restrict__ Kg,
                                         const float* __restrict__ Vg,
                                         long gbase, __bf16* ksb, __bf16* vsb,
                                         int tid) {
#pragma unroll
  for (int i = 0; i < 4; ++i) {
    int e  = i * (NTHREADS * 2) + tid * 2;   // 0..2046 step 2
    int kk = e >> 6, d = e & 63;             // d even
    const float2 k2 = *(const float2*)&Kg[gbase + e];
    const float2 v2 = *(const float2*)&Vg[gbase + e];
    *(unsigned int*)&ksb[e] = bfpack(k2.x, k2.y);   // [key][d]
    vsb[d * KTILE + kk]       = (__bf16)v2.x;       // [d][key] transposed
    vsb[(d + 1) * KTILE + kk] = (__bf16)v2.y;
  }
}

__global__ __launch_bounds__(NTHREADS)
void fa_causal_bf16_wmma(const float* __restrict__ Vg,
                         const float* __restrict__ Kg,
                         const float* __restrict__ Qg,
                         float* __restrict__ Og)
{
  const int tid    = threadIdx.x;
  const int wave   = tid >> 5;
  const int lane   = tid & 31;
  const int lane16 = lane & 15;
  const int hi     = lane >> 4;
  const int q0     = blockIdx.x * QTILE;
  const long base  = (long)blockIdx.y * (long)L_SEQ * D_HEAD;

  // LDS: 16 + 8 + 8 + 16 = 48 KB of 320 KB/WGP.
  __shared__ __bf16 qs[QTILE * D_HEAD];          // Q bf16 [row][d]
  __shared__ __bf16 ks[2][KTILE * D_HEAD];       // K bf16 [key][d], double-buffered
  __shared__ __bf16 vs[2][D_HEAD * KTILE];       // V bf16 [d][key], double-buffered
  __shared__ float  ss[8][16 * KTILE];           // per-wave S scratch [row][key]

  // ---- Q tile: 128x64 fp32 -> bf16 LDS ----
#pragma unroll
  for (int i = 0; i < 16; ++i) {
    int e = i * (NTHREADS * 2) + tid * 2;
    const float2 q2 = *(const float2*)&Qg[base + (long)q0 * D_HEAD + e];
    *(unsigned int*)&qs[e] = bfpack(q2.x, q2.y);
  }
  // first K/V tile into buffer 0
  stage_kv(Kg, Vg, base, ks[0], vs[0], tid);
  __syncthreads();

  // ---- Q A-fragments (16x32 bf16, two d-halves) ----
  const int qrow = wave * 16 + lane16;
  Frag16 qf[2];
#pragma unroll
  for (int hf = 0; hf < 2; ++hf)
#pragma unroll
    for (int i = 0; i < 8; ++i) {
      int kb = hf * 32 + hi * 16 + 2 * i;
      qf[hf].u[i] = *(const unsigned int*)&qs[qrow * D_HEAD + kb];
    }

  v8f o[4] = {v8f{}, v8f{}, v8f{}, v8f{}};
  float m = -1e30f, l = 0.0f;                    // per-lane (row = lane16), base-2 domain

  const int my_row       = q0 + wave * 16 + lane16;
  const int rowg         = q0 + wave * 16 + hi * 8;   // + r = row of S VGPR r
  const int wave_max_row = q0 + wave * 16 + 15;
  const int diag_j       = (q0 + wave * 16) & ~(KTILE - 1);  // only tile needing mask
  const int jmax         = q0 + QTILE;

  for (int j = 0; j < jmax; j += KTILE) {
    const int cur = (j >> 5) & 1;
    if (j + KTILE < jmax) {                      // block-uniform: prefetch next tile
      stage_kv(Kg, Vg, base + (long)(j + KTILE) * D_HEAD, ks[cur ^ 1], vs[cur ^ 1], tid);
      __builtin_prefetch(&Kg[base + (long)(j + 2 * KTILE) * D_HEAD + lane * 16], 0, 3);
      __builtin_prefetch(&Vg[base + (long)(j + 2 * KTILE) * D_HEAD + lane * 16], 0, 3);
    }

    if (j <= wave_max_row) {                     // wave-uniform; EXEC all-ones at WMMAs
      const __bf16* ksb = ks[cur];
      const __bf16* vsb = vs[cur];

      // ---- S = Q K^T : 4 WMMAs ----
      v8f s[2];
#pragma unroll
      for (int kk = 0; kk < 2; ++kk) {
        v8f acc = v8f{};
#pragma unroll
        for (int hf = 0; hf < 2; ++hf) {
          Frag16 bfrag;   // B layout: lane N = lane16; v[i] = K pair (hi*16 + 2i, +1)
          const __bf16* kr = &ksb[(kk * 16 + lane16) * D_HEAD + hf * 32 + hi * 16];
#pragma unroll
          for (int i = 0; i < 8; ++i)
            bfrag.u[i] = *(const unsigned int*)&kr[2 * i];
          acc = __builtin_amdgcn_wmma_f32_16x16x32_bf16(
                    false, qf[hf].bf, false, bfrag.bf, (short)0, acc, false, false);
        }
        s[kk] = acc;
      }

      // ---- scale (base-2) + spill to LDS; mask only on the diagonal tile ----
      float* sw = ss[wave];
      if (j == diag_j) {
#pragma unroll
        for (int kk = 0; kk < 2; ++kk) {
          const int keyg = j + kk * 16 + lane16;
#pragma unroll
          for (int r = 0; r < 8; ++r) {
            float v = s[kk][r] * SCALE_LOG2E;
            if (keyg > rowg + r) v = -1e30f;     // causal
            sw[(hi * 8 + r) * KTILE + kk * 16 + lane16] = v;
          }
        }
      } else {
#pragma unroll
        for (int kk = 0; kk < 2; ++kk)
#pragma unroll
          for (int r = 0; r < 8; ++r)
            sw[(hi * 8 + r) * KTILE + kk * 16 + lane16] = s[kk][r] * SCALE_LOG2E;
      }

      // ---- read back row-per-lane (P^T B-fragment key order) ----
      float sv[16];
#pragma unroll
      for (int i = 0; i < 16; ++i)
        sv[i] = sw[lane16 * KTILE + hi * 16 + i];

      // ---- online softmax (base-2), per-lane scalars ----
      float mx = sv[0];
#pragma unroll
      for (int i = 1; i < 16; ++i) mx = fmaxf(mx, sv[i]);
      mx = fmaxf(mx, swap16_f(mx));
      const float mn    = fmaxf(m, mx);
      const float alpha = EXP2(m - mn);
      m = mn;
      float p[16], sum = 0.0f;
#pragma unroll
      for (int i = 0; i < 16; ++i) { p[i] = EXP2(sv[i] - mn); sum += p[i]; }
      sum += swap16_f(sum);
      l = l * alpha + sum;

      // ---- rescale O^T (per-lane scalar alpha) ----
#pragma unroll
      for (int t = 0; t < 4; ++t)
#pragma unroll
        for (int r = 0; r < 8; ++r)
          o[t][r] *= alpha;

      // ---- P^T B-fragment straight from p[] ----
      Frag16 pf;
#pragma unroll
      for (int i = 0; i < 8; ++i)
        pf.u[i] = bfpack(p[2 * i], p[2 * i + 1]);

      // ---- O^T += V^T P^T : 4 WMMAs ----
#pragma unroll
      for (int t = 0; t < 4; ++t) {
        Frag16 vf;     // A = V^T: v0..3 keys hi*16+{0..7}, v4..7 keys hi*16+8+{0..7}
        const __bf16* vr = &vsb[(t * 16 + lane16) * KTILE];
#pragma unroll
        for (int i = 0; i < 4; ++i) {
          vf.u[i]     = *(const unsigned int*)&vr[hi * 16 + 2 * i];
          vf.u[i + 4] = *(const unsigned int*)&vr[hi * 16 + 8 + 2 * i];
        }
        o[t] = __builtin_amdgcn_wmma_f32_16x16x32_bf16(
                   false, vf.bf, false, pf.bf, (short)0, o[t], false, false);
      }
    }
    __syncthreads();   // compute on buf[cur] done; staging of buf[cur^1] visible
  }

  // ---- epilogue: lane owns row my_row, d = 16t + hi*8 + {0..7} ----
  const float inv = 1.0f / l;
  const long ob = base + (long)my_row * D_HEAD + hi * 8;
#pragma unroll
  for (int t = 0; t < 4; ++t) {
    float4 lo = make_float4(o[t][0] * inv, o[t][1] * inv, o[t][2] * inv, o[t][3] * inv);
    float4 hh = make_float4(o[t][4] * inv, o[t][5] * inv, o[t][6] * inv, o[t][7] * inv);
    *(float4*)&Og[ob + 16 * t]     = lo;
    *(float4*)&Og[ob + 16 * t + 4] = hh;
  }
}

extern "C" void kernel_launch(void* const* d_in, const int* in_sizes, int n_in,
                              void* d_out, int out_size, void* d_ws, size_t ws_size,
                              hipStream_t stream) {
  const float* x = (const float*)d_in[0];   // V
  const float* B = (const float*)d_in[1];   // K
  const float* C = (const float*)d_in[2];   // Q
  float* out = (float*)d_out;
  (void)in_sizes; (void)n_in; (void)out_size; (void)d_ws; (void)ws_size;

  dim3 grid(L_SEQ / QTILE, 2 * 8);          // (query tiles, batch*heads)
  dim3 block(NTHREADS);
  fa_causal_bf16_wmma<<<grid, block, 0, stream>>>(x, B, C, out);
}
